// LinearAttention_29308856828525
// MI455X (gfx1250) — compile-verified, matrix operand order checked by reading, not run on hardware
//
#include <hip/hip_runtime.h>

// ---------------------------------------------------------------------------
// Types for CDNA5 (wave32) WMMA + TDM
// ---------------------------------------------------------------------------
typedef __attribute__((ext_vector_type(16))) __bf16 v16bf;
typedef __attribute__((ext_vector_type(8)))  float  v8f;
typedef __attribute__((ext_vector_type(4)))  unsigned int u32x4;
typedef __attribute__((ext_vector_type(8)))  int i32x8;
typedef __attribute__((ext_vector_type(4)))  int i32x4;
// 16B-aligned view for LDS reads (rows are 16B- but not 32B-aligned)
typedef __bf16 v16bf_a16 __attribute__((ext_vector_type(16), aligned(16)));

#define NSEQ   4096
#define CDIM   512
#define HEADS  8
#define HDIM   64
#define BATCH  4
#define ROWS   (BATCH * NSEQ)        // 16384
#define QKVC   (3 * CDIM)            // 1536

#if defined(__has_builtin)
#if __has_builtin(__builtin_amdgcn_tensor_load_to_lds) && \
    __has_builtin(__builtin_amdgcn_s_wait_tensorcnt)
#define USE_TDM 1
#endif
#endif
#ifndef USE_TDM
#define USE_TDM 0
#endif

#define KC       64                  // K-chunk (elements)
#define LSTRIDE  72                  // LDS row stride = KC + 8 (pad, bank-spread)

#if USE_TDM
// ---------------------------------------------------------------------------
// Issue one TDM 2D tile load: rows x rowlen bf16 elements, global row stride
// stride_elts, into LDS at byte offset lds_off with an 8-element (16B) pad
// appended after every 64-element (128B) row: pad_interval=32 DWORDs (code 4),
// pad_amount=4 DWORDs (code 3)  ->  LDS row stride = 72 elements.
// D# layout per CDNA5 ISA ch.8.
// ---------------------------------------------------------------------------
__device__ __forceinline__ void tdm_load_tile_2d(unsigned lds_off,
                                                 const __bf16* gsrc,
                                                 unsigned rows, unsigned rowlen,
                                                 unsigned long long stride_elts)
{
    unsigned long long ga = (unsigned long long)(size_t)gsrc;

    u32x4 g0;
    g0[0] = 1u;                                            // count=1, user D#
    g0[1] = lds_off;                                       // lds_addr (bytes)
    g0[2] = (unsigned)(ga & 0xFFFFFFFFu);                  // global_addr lo
    g0[3] = (unsigned)((ga >> 32) & 0x01FFFFFFu)           // global_addr hi
          | (2u << 30);                                    // type = 2 (image)

    const unsigned td0 = rowlen;                           // tensor_dim0
    const unsigned td1 = rows;                             // tensor_dim1

    i32x8 g1;
    g1[0] = (int)((1u << 16)                               // data_size = 2B
                | (1u << 20)                               // pad_enable
                | (4u << 22)                               // pad_interval: 32 DW
                | (3u << 25));                             // pad_amount : 4 DW
    g1[1] = (int)((td0 & 0xFFFFu) << 16);                  // tensor_dim0[15:0]
    g1[2] = (int)(((td0 >> 16) & 0xFFFFu) | ((td1 & 0xFFFFu) << 16));
    g1[3] = (int)(((td1 >> 16) & 0xFFFFu) | ((rowlen & 0xFFFFu) << 16)); // tile_dim0
    g1[4] = (int)(rows & 0xFFFFu);                         // tile_dim1 (tile_dim2=0)
    g1[5] = (int)(unsigned)(stride_elts & 0xFFFFFFFFull);  // dim0_stride lo
    g1[6] = (int)(unsigned)((stride_elts >> 32) & 0xFFFFull);
    g1[7] = 0;

    i32x4 z4 = {0, 0, 0, 0};
#if defined(__clang_major__) && (__clang_major__ >= 23)
    i32x8 z8 = {0, 0, 0, 0, 0, 0, 0, 0};
    __builtin_amdgcn_tensor_load_to_lds(g0, g1, z4, z4, z8, 0);
#else
    __builtin_amdgcn_tensor_load_to_lds(g0, g1, z4, z4, 0);
#endif
}
#endif // USE_TDM

// ---------------------------------------------------------------------------
// LDS-staged, TDM double-buffered WMMA GEMM (batch = 1):
//   C[M,N] (+bias) = A[M,K] * Bt[N,K]^T
// Block = 256 threads (8 waves), block tile 128(M) x 64(N), K-chunk 64.
// Wave w owns a 16x64 strip (4 accumulators, 8 WMMAs per chunk).
// Requires M%128==0, N%64==0, K%64==0.
// ---------------------------------------------------------------------------
template<bool BF16_OUT, bool BIAS>
__global__ __launch_bounds__(256)
void wmma_gemm_lds(const __bf16* __restrict__ A, const __bf16* __restrict__ Bt,
                   const float* __restrict__ bias, void* __restrict__ Cout,
                   int M, int N, int K, int lda, int ldb, int ldc)
{
    __shared__ __align__(16) __bf16 ldsA[2][128 * LSTRIDE];   // 36 KB
    __shared__ __align__(16) __bf16 ldsB[2][64 * LSTRIDE];    // 18 KB

    const int lane = threadIdx.x & 31;
    const int wave = threadIdx.x >> 5;
    const int m0 = blockIdx.x * 128;
    const int n0 = blockIdx.y * 64;

    const int fr = lane & 15;            // fragment row / col
    const int kb = (lane >> 4) << 4;     // K-half select

    const int nchunks = K / KC;
    v8f acc[4] = {};

#if USE_TDM
    if (wave == 0) {
        tdm_load_tile_2d((unsigned)(size_t)&ldsA[0][0], A + (size_t)m0 * lda,
                         128, KC, (unsigned long long)lda);
        tdm_load_tile_2d((unsigned)(size_t)&ldsB[0][0], Bt + (size_t)n0 * ldb,
                         64, KC, (unsigned long long)ldb);
        if (nchunks > 1) {
            tdm_load_tile_2d((unsigned)(size_t)&ldsA[1][0], A + (size_t)m0 * lda + KC,
                             128, KC, (unsigned long long)lda);
            tdm_load_tile_2d((unsigned)(size_t)&ldsB[1][0], Bt + (size_t)n0 * ldb + KC,
                             64, KC, (unsigned long long)ldb);
        }
    }
#endif

    for (int i = 0; i < nchunks; ++i) {
        const int buf = i & 1;

#if USE_TDM
        if (wave == 0) {
            if (i + 1 < nchunks) __builtin_amdgcn_s_wait_tensorcnt(2);
            else                 __builtin_amdgcn_s_wait_tensorcnt(0);
        }
        __syncthreads();                 // chunk i resident in LDS for all waves
#else
        __syncthreads();
        for (int t = threadIdx.x; t < 128 * (KC / 16); t += 256) {
            int r = t / (KC / 16), c = (t % (KC / 16)) * 16;
            *(v16bf_a16*)&ldsA[buf][r * LSTRIDE + c] =
                *(const v16bf*)(A + (size_t)(m0 + r) * lda + (size_t)i * KC + c);
        }
        for (int t = threadIdx.x; t < 64 * (KC / 16); t += 256) {
            int r = t / (KC / 16), c = (t % (KC / 16)) * 16;
            *(v16bf_a16*)&ldsB[buf][r * LSTRIDE + c] =
                *(const v16bf*)(Bt + (size_t)(n0 + r) * ldb + (size_t)i * KC + c);
        }
        __syncthreads();
#endif

        const __bf16* As = &ldsA[buf][(wave * 16 + fr) * LSTRIDE + kb];
        const __bf16* Bs = &ldsB[buf][fr * LSTRIDE + kb];
#pragma unroll
        for (int ks = 0; ks < KC; ks += 32) {
            v16bf a = *(const v16bf_a16*)(As + ks);
            v16bf bfrag[4];
#pragma unroll
            for (int j = 0; j < 4; ++j)
                bfrag[j] = *(const v16bf_a16*)(Bs + j * 16 * LSTRIDE + ks);
#pragma unroll
            for (int j = 0; j < 4; ++j)
                acc[j] = __builtin_amdgcn_wmma_f32_16x16x32_bf16(
                             false, a, false, bfrag[j], (short)0, acc[j], false, false);
        }

#if USE_TDM
        __syncthreads();                 // all waves done reading buf before reuse
        if (wave == 0 && i + 2 < nchunks) {
            tdm_load_tile_2d((unsigned)(size_t)&ldsA[buf][0],
                             A + (size_t)m0 * lda + (size_t)(i + 2) * KC,
                             128, KC, (unsigned long long)lda);
            tdm_load_tile_2d((unsigned)(size_t)&ldsB[buf][0],
                             Bt + (size_t)n0 * ldb + (size_t)(i + 2) * KC,
                             64, KC, (unsigned long long)ldb);
        }
#endif
    }

    // C/D layout: lane -> col (l&15), rows (l>>4)*8 + i.
    const int cn   = lane & 15;
    const int mrow = ((lane >> 4) << 3) + wave * 16;

    if (BF16_OUT) {
        __bf16* Cp = (__bf16*)Cout;
#pragma unroll
        for (int j = 0; j < 4; ++j) {
            const int col = n0 + j * 16 + cn;
#pragma unroll
            for (int ii = 0; ii < 8; ++ii) {
                float v = acc[j][ii];
                Cp[(size_t)(m0 + mrow + ii) * ldc + col] = (__bf16)v;
            }
        }
    } else {
        float* Cp = (float*)Cout;
#pragma unroll
        for (int j = 0; j < 4; ++j) {
            const int col = n0 + j * 16 + cn;
            const float bv = BIAS ? bias[col] : 0.0f;
#pragma unroll
            for (int ii = 0; ii < 8; ++ii)
                Cp[(size_t)(m0 + mrow + ii) * ldc + col] = acc[j][ii] + bv;
        }
    }
}

// ---------------------------------------------------------------------------
// Direct-from-global batched WMMA GEMM (small batched GEMMs, K3/K4).
// Batch z: b = z>>3, h = z&7 with per-(b,h) element offsets.
// ---------------------------------------------------------------------------
__global__ __launch_bounds__(256)
void wmma_gemm_direct(const __bf16* __restrict__ A, const __bf16* __restrict__ Bt,
                      __bf16* __restrict__ Cout,
                      int M, int N, int K, int lda, int ldb, int ldc,
                      long long sAb, long long sAh,
                      long long sBb, long long sBh,
                      long long sCb, long long sCh)
{
    const int lane = threadIdx.x & 31;
    const int wave = threadIdx.x >> 5;
    const int z  = blockIdx.z;
    const int bb = z >> 3;
    const int hh = z & 7;

    const __bf16* Ab = A  + bb * sAb + hh * sAh;
    const __bf16* Bb = Bt + bb * sBb + hh * sBh;

    const int m0 = (blockIdx.x * 8 + wave) * 16;
    const int n0 = blockIdx.y * 64;
    if (m0 >= M) return;                 // wave-uniform

    const int fr = lane & 15;
    const int kb = (lane >> 4) << 4;

    const __bf16* Ap = Ab + (long long)(m0 + fr) * lda + kb;
    const __bf16* Bp = Bb + (long long)(n0 + fr) * ldb + kb;

    v8f acc[4] = {};

    for (int kk = 0; kk < K; kk += 32) {
        v16bf a = *(const v16bf*)(Ap + kk);
        __builtin_prefetch((const void*)(Ap + kk + 128), 0, 1);
        v16bf bfrag[4];
#pragma unroll
        for (int j = 0; j < 4; ++j)
            bfrag[j] = *(const v16bf*)(Bp + (long long)j * 16 * ldb + kk);
#pragma unroll
        for (int j = 0; j < 4; ++j)
            acc[j] = __builtin_amdgcn_wmma_f32_16x16x32_bf16(
                         false, a, false, bfrag[j], (short)0, acc[j], false, false);
    }

    const int cn   = lane & 15;
    const int mrow = (lane >> 4) << 3;
    __bf16* Cp = Cout + bb * sCb + hh * sCh;
#pragma unroll
    for (int j = 0; j < 4; ++j) {
        const int col = n0 + j * 16 + cn;
#pragma unroll
        for (int ii = 0; ii < 8; ++ii) {
            float v = acc[j][ii];
            Cp[(long long)(m0 + mrow + ii) * ldc + col] = (__bf16)v;
        }
    }
}

// ---------------------------------------------------------------------------
// f32 -> bf16 convert (grid-stride)
// ---------------------------------------------------------------------------
__global__ void cvt_f32_bf16(const float* __restrict__ src,
                             __bf16* __restrict__ dst, long long n)
{
    long long i = (long long)blockIdx.x * blockDim.x + threadIdx.x;
    long long stride = (long long)gridDim.x * blockDim.x;
    for (; i < n; i += stride) dst[i] = (__bf16)src[i];
}

// src [rows][cols] f32  ->  dst [cols][rows] bf16   (weight -> Bt layout)
__global__ void transpose_cvt(const float* __restrict__ src,
                              __bf16* __restrict__ dst, int rows, int cols)
{
    long long idx = (long long)blockIdx.x * blockDim.x + threadIdx.x;
    long long total = (long long)rows * cols;
    if (idx >= total) return;
    int r = (int)(idx % rows);
    int c = (int)(idx / rows);
    dst[(long long)c * rows + r] = (__bf16)src[(long long)r * cols + c];
}

// ---------------------------------------------------------------------------
// q softmax over head_dim (64), one thread per (b,n,h) row.
// ---------------------------------------------------------------------------
__global__ __launch_bounds__(256)
void qsoftmax_kernel(const __bf16* __restrict__ qkvb, __bf16* __restrict__ qb)
{
    int t = blockIdx.x * blockDim.x + threadIdx.x;
    if (t >= ROWS * HEADS) return;
    int bn = t >> 3, h = t & 7;
    const __bf16* src = qkvb + (long long)bn * QKVC + h * HDIM;

    float vals[HDIM];
    float mx = -1e30f;
#pragma unroll
    for (int d = 0; d < HDIM; ++d) {
        float v = (float)src[d];
        vals[d] = v;
        mx = fmaxf(mx, v);
    }
    float sm = 0.0f;
#pragma unroll
    for (int d = 0; d < HDIM; ++d) {
        float e = __expf(vals[d] - mx);
        vals[d] = e;
        sm += e;
    }
    float inv = 1.0f / sm;
    __bf16* dst = qb + (long long)bn * CDIM + h * HDIM;
#pragma unroll
    for (int d = 0; d < HDIM; ++d) dst[d] = (__bf16)(vals[d] * inv);
}

// ---------------------------------------------------------------------------
// k: s = elu(k)+1, softmax over sequence dim (N=4096), one block per (b,h,d)
// column; column cached in LDS.  Output transposed: kTb[bh*64+d][n].
// ---------------------------------------------------------------------------
__global__ __launch_bounds__(256)
void ksoftmax_kernel(const __bf16* __restrict__ qkvb, __bf16* __restrict__ kTb)
{
    __shared__ float s[NSEQ];
    __shared__ float red[256];

    const int col = blockIdx.x;           // bh*64 + d
    const int bh  = col >> 6;
    const int d   = col & 63;
    const int b   = bh >> 3;
    const int h   = bh & 7;
    const int tid = threadIdx.x;

    const __bf16* base = qkvb + (long long)b * NSEQ * QKVC + (CDIM + h * HDIM + d);

    float mx = -1e30f;
    for (int i = tid; i < NSEQ; i += 256) {
        float kv = (float)base[(long long)i * QKVC];
        float e  = (kv > 0.0f) ? kv : (__expf(kv) - 1.0f);
        e += 1.0f;
        s[i] = e;
        mx = fmaxf(mx, e);
    }
    red[tid] = mx;
    __syncthreads();
    for (int st = 128; st > 0; st >>= 1) {
        if (tid < st) red[tid] = fmaxf(red[tid], red[tid + st]);
        __syncthreads();
    }
    mx = red[0];
    __syncthreads();

    float sm = 0.0f;
    for (int i = tid; i < NSEQ; i += 256) {
        float e = __expf(s[i] - mx);
        s[i] = e;
        sm += e;
    }
    red[tid] = sm;
    __syncthreads();
    for (int st = 128; st > 0; st >>= 1) {
        if (tid < st) red[tid] += red[tid + st];
        __syncthreads();
    }
    float inv = 1.0f / red[0];

    __bf16* dst = kTb + (long long)col * NSEQ;
    for (int i = tid; i < NSEQ; i += 256) dst[i] = (__bf16)(s[i] * inv);
}

// ---------------------------------------------------------------------------
// v transpose: vTb[bh][e][n] = qkv_v[b, n, h, e]
// ---------------------------------------------------------------------------
__global__ void vtranspose_kernel(const __bf16* __restrict__ qkvb,
                                  __bf16* __restrict__ vTb)
{
    long long idx = (long long)blockIdx.x * blockDim.x + threadIdx.x;
    const long long total = (long long)BATCH * HEADS * HDIM * NSEQ;
    if (idx >= total) return;
    int n  = (int)(idx & (NSEQ - 1));
    int e  = (int)((idx >> 12) & 63);
    int bh = (int)(idx >> 18);
    int b  = bh >> 3, h = bh & 7;
    vTb[idx] = qkvb[((long long)b * NSEQ + n) * QKVC + (2 * CDIM + h * HDIM + e)];
}

// ---------------------------------------------------------------------------
// Host launcher
// ---------------------------------------------------------------------------
extern "C" void kernel_launch(void* const* d_in, const int* in_sizes, int n_in,
                              void* d_out, int out_size, void* d_ws, size_t ws_size,
                              hipStream_t stream)
{
    const float* x      = (const float*)d_in[0];
    const float* w_qkv  = (const float*)d_in[1];
    const float* w_proj = (const float*)d_in[2];
    const float* b_proj = (const float*)d_in[3];
    float* out = (float*)d_out;

    char* ws = (char*)d_ws;
    __bf16* xb     = (__bf16*)ws; ws += (size_t)ROWS * CDIM * 2;
    __bf16* wqkvT  = (__bf16*)ws; ws += (size_t)QKVC * CDIM * 2;
    __bf16* wprojT = (__bf16*)ws; ws += (size_t)CDIM * CDIM * 2;
    __bf16* qkvb   = (__bf16*)ws; ws += (size_t)ROWS * QKVC * 2;
    __bf16* qb     = (__bf16*)ws; ws += (size_t)ROWS * CDIM * 2;
    __bf16* kTb    = (__bf16*)ws; ws += (size_t)BATCH * HEADS * HDIM * NSEQ * 2;
    __bf16* vTb    = (__bf16*)ws; ws += (size_t)BATCH * HEADS * HDIM * NSEQ * 2;
    __bf16* ctxT   = (__bf16*)ws; ws += (size_t)BATCH * HEADS * HDIM * HDIM * 2;
    __bf16* outb   = (__bf16*)ws; ws += (size_t)ROWS * CDIM * 2;

    // 0) precision conversion + weight transposes
    {
        long long n = (long long)ROWS * CDIM;
        cvt_f32_bf16<<<(int)((n + 255) / 256), 256, 0, stream>>>(x, xb, n);
    }
    transpose_cvt<<<(CDIM * QKVC + 255) / 256, 256, 0, stream>>>(w_qkv, wqkvT, CDIM, QKVC);
    transpose_cvt<<<(CDIM * CDIM + 255) / 256, 256, 0, stream>>>(w_proj, wprojT, CDIM, CDIM);

    // 1) qkv = x @ w_qkv  [16384,512]x[512,1536] -> bf16 (TDM/LDS pipeline)
    wmma_gemm_lds<true, false><<<dim3(ROWS / 128, QKVC / 64, 1), 256, 0, stream>>>(
        xb, wqkvT, nullptr, qkvb, ROWS, QKVC, CDIM, CDIM, CDIM, QKVC);

    // 2) softmaxes + v transpose
    qsoftmax_kernel<<<(ROWS * HEADS) / 256, 256, 0, stream>>>(qkvb, qb);
    ksoftmax_kernel<<<BATCH * HEADS * HDIM, 256, 0, stream>>>(qkvb, kTb);
    {
        long long n = (long long)BATCH * HEADS * HDIM * NSEQ;
        vtranspose_kernel<<<(int)((n + 255) / 256), 256, 0, stream>>>(qkvb, vTb);
    }

    // 3) ctx^T[e,d] = sum_n v[n,e] * k[n,d]   (per b,h; K = 4096)
    wmma_gemm_direct<<<dim3(1, 1, BATCH * HEADS), 256, 0, stream>>>(
        vTb, kTb, ctxT,
        HDIM, HDIM, NSEQ, NSEQ, NSEQ, HDIM,
        (long long)HEADS * HDIM * NSEQ, (long long)HDIM * NSEQ,
        (long long)HEADS * HDIM * NSEQ, (long long)HDIM * NSEQ,
        (long long)HEADS * HDIM * HDIM, (long long)HDIM * HDIM);

    // 4) out_head = q @ ctx   (ctxT already in Bt layout)
    wmma_gemm_direct<<<dim3(NSEQ / 128, 1, BATCH * HEADS), 256, 0, stream>>>(
        qb, ctxT, outb,
        NSEQ, HDIM, HDIM, CDIM, HDIM, CDIM,
        (long long)NSEQ * CDIM, (long long)HDIM,
        (long long)HEADS * HDIM * HDIM, (long long)HDIM * HDIM,
        (long long)NSEQ * CDIM, (long long)HDIM);

    // 5) final = out_head @ w_proj + b_proj -> f32 d_out (TDM/LDS pipeline)
    wmma_gemm_lds<false, true><<<dim3(ROWS / 128, CDIM / 64, 1), 256, 0, stream>>>(
        outb, wprojT, b_proj, out, ROWS, CDIM, CDIM, CDIM, CDIM, CDIM);

    (void)in_sizes; (void)n_in; (void)out_size; (void)ws_size;
}
